// GKT_88338887344567
// MI455X (gfx1250) — compile-verified, matrix operand order checked by reading, not run on hardware
//
#include <hip/hip_runtime.h>
#include <hip/hip_bf16.h>

#define B_   32
#define T_   64
#define NQ_  128
#define H_   128
#define G3_  384   // 3*H

typedef __bf16 bf16;
typedef __attribute__((ext_vector_type(16))) __bf16        v16bf;
typedef __attribute__((ext_vector_type(8)))  float         v8f;
typedef __attribute__((ext_vector_type(4)))  unsigned int  u32x4;

union FragBF { v16bf v; u32x4 q[2]; };

__device__ __forceinline__ float sigmoidf_(float x) { return 1.0f / (1.0f + __expf(-x)); }

#define WMMA_BF16(A, Bm, C) \
  __builtin_amdgcn_wmma_f32_16x16x32_bf16(false, (A), false, (Bm), (short)0, (C), false, false)

// ---------------------------------------------------------------------------
// Init: h[:,0] = init_h broadcast; convert recurrent weights to bf16 in ws.
// ---------------------------------------------------------------------------
__global__ __launch_bounds__(256)
void gkt_init(const float* __restrict__ init_h,
              const float* __restrict__ w_ih,
              const float* __restrict__ w_hh,
              float* __restrict__ hout,
              bf16* __restrict__ whh_bf,
              bf16* __restrict__ wh_bf) {
  int idx = blockIdx.x * blockDim.x + threadIdx.x;
  if (idx < B_ * NQ_ * H_) {
    int b = idx >> 14;            // / (NQ_*H_)
    int rest = idx & 16383;
    hout[(size_t)(b * (T_ + 1)) * (NQ_ * H_) + rest] = init_h[rest];
  }
  if (idx < G3_ * H_) {
    int g = idx >> 7, k = idx & 127;
    whh_bf[idx] = (bf16)w_hh[idx];                       // [384][128]
    wh_bf[idx]  = (bf16)w_ih[g * (2 * H_) + H_ + k];     // second-half cols of w_ih
  }
}

// ---------------------------------------------------------------------------
// Kernel A: gi[b,n,:] = b_ih + ht[b,n,:] @ w_ih[:,H:].T    (4096 x 384 x K=128)
// One block = one 16-row M tile; 8 waves x 3 N-tiles; bf16 WMMA, f32 acc.
// ---------------------------------------------------------------------------
__global__ __launch_bounds__(256)
void gkt_gemm_gi(const float* __restrict__ h,
                 const bf16* __restrict__ wh_bf,
                 const float* __restrict__ b_ih,
                 float* __restrict__ gi,
                 int t) {
  __shared__ bf16 Abf[16 * 136];   // 16 rows, pad to 136 to spread LDS banks
  const int tid = threadIdx.x;
  const int mt  = blockIdx.x;            // 0..255
  const int b   = mt >> 3;
  const int n0  = (mt & 7) * 16;

  { // stage + convert A tile (16x128 f32 -> bf16)
    int r  = tid >> 4;
    int k8 = (tid & 15) * 8;
    const float* src = h + ((size_t)(b * (T_ + 1) + t) * NQ_ + n0 + r) * H_ + k8;
    bf16* dst = Abf + r * 136 + k8;
    #pragma unroll
    for (int i = 0; i < 8; ++i) dst[i] = (bf16)src[i];
  }
  __syncthreads();

  const int lane = tid & 31;
  const int wv   = tid >> 5;             // 0..7
  const int ln15 = lane & 15;
  const int e0   = (lane >> 4) * 8;      // K sub-offset per lane half
  const int g0   = 48 * wv;              // 3 N-tiles per wave

  const bf16* wp0 = wh_bf + (size_t)(g0      + ln15) * H_;
  const bf16* wp1 = wh_bf + (size_t)(g0 + 16 + ln15) * H_;
  const bf16* wp2 = wh_bf + (size_t)(g0 + 32 + ln15) * H_;

  v8f c0 = {}, c1 = {}, c2 = {};
  #pragma unroll
  for (int ks = 0; ks < 4; ++ks) {
    const int k0 = ks * 32;
    FragBF a, f0, f1, f2;
    a.q[0]  = *(const u32x4*)(Abf + ln15 * 136 + k0 + e0);
    a.q[1]  = *(const u32x4*)(Abf + ln15 * 136 + k0 + e0 + 16);
    f0.q[0] = *(const u32x4*)(wp0 + k0 + e0);
    f0.q[1] = *(const u32x4*)(wp0 + k0 + e0 + 16);
    f1.q[0] = *(const u32x4*)(wp1 + k0 + e0);
    f1.q[1] = *(const u32x4*)(wp1 + k0 + e0 + 16);
    f2.q[0] = *(const u32x4*)(wp2 + k0 + e0);
    f2.q[1] = *(const u32x4*)(wp2 + k0 + e0 + 16);
    c0 = WMMA_BF16(a.v, f0.v, c0);
    c1 = WMMA_BF16(a.v, f1.v, c1);
    c2 = WMMA_BF16(a.v, f2.v, c2);
  }

  // D layout: VGPR v -> M = (lane>>4)*8 + v ; N = lane&15
  const int row0 = mt * 16 + (lane >> 4) * 8;
  const float bv0 = b_ih[g0      + ln15];
  const float bv1 = b_ih[g0 + 16 + ln15];
  const float bv2 = b_ih[g0 + 32 + ln15];
  #pragma unroll
  for (int v = 0; v < 8; ++v) {
    float* grow = gi + (size_t)(row0 + v) * G3_;
    grow[g0      + ln15] = c0[v] + bv0;
    grow[g0 + 16 + ln15] = c1[v] + bv1;
    grow[g0 + 32 + ln15] = c2[v] + bv2;
  }
}

// ---------------------------------------------------------------------------
// Kernel B: one persistent workgroup per time step.
//   prologue: per-batch MLP correction into gi at row q[b,t]
//   scan n=0..127: gh = carry @ w_hh.T (WMMA, weights pinned in VGPRs),
//                  GRU elementwise, write h[:,t+1,n,:] and y[:,t,n].
// ---------------------------------------------------------------------------
__global__ __launch_bounds__(512)
void gkt_scan(const int* __restrict__ qk, const int* __restrict__ rk,
              const float* __restrict__ x_emb,
              const float* __restrict__ w1, const float* __restrict__ b1,
              const float* __restrict__ w2, const float* __restrict__ b2,
              const float* __restrict__ w_ih, const float* __restrict__ b_hh,
              const float* __restrict__ bias, const float* __restrict__ out_w,
              const bf16* __restrict__ whh_bf,
              float* __restrict__ gi,
              float* __restrict__ hout, float* __restrict__ yout,
              int t) {
  __shared__ float gh_ls[32 * 388];       // gates (f32), also MLP scratch
  __shared__ bf16  hbf[32 * 136];         // bf16 mirror of GRU carry (A matrix)
  __shared__ float y_acc[32];
  __shared__ float bias_s[NQ_], outw_s[H_], bhh_s[G3_];
  __shared__ int   qt_s[32];

  const int tid = threadIdx.x;

  if (tid < 128) { bias_s[tid] = bias[tid]; outw_s[tid] = out_w[tid]; }
  if (tid < 384) bhh_s[tid] = b_hh[tid];
  if (tid < 32)  { qt_s[tid] = qk[tid * T_ + t]; y_acc[tid] = 0.f; }
  for (int i = tid; i < 32 * 136; i += 512) hbf[i] = (bf16)0.f;
  __syncthreads();

  // ---- MLP correction: gi[b, q[b,t], :] += w_ih[:, :H] @ mlp(ht[b,qt], xt[b])
  const int bb = tid >> 4;                 // 0..31
  float* zbuf = gh_ls;                     // [32][256]
  float* hid  = gh_ls + 32 * 256;          // [32][128]
  float* mv   = gh_ls;                     // [32][128] (reuses z region)
  {
    const int qt = qt_s[bb];
    const int xi = qt + NQ_ * rk[bb * T_ + t];
    const float* hrow = hout + ((size_t)(bb * (T_ + 1) + t) * NQ_ + qt) * H_;
    const float* xrow = x_emb + (size_t)xi * H_;
    const int c0 = (tid & 15) * 16;
    #pragma unroll
    for (int i = 0; i < 16; ++i) {
      int c = c0 + i;
      zbuf[bb * 256 + c] = (c < H_) ? hrow[c] : xrow[c - H_];
    }
  }
  __syncthreads();
  {
    const int j0 = (tid & 15) * 8;
    const float* zr = zbuf + bb * 256;
    #pragma unroll
    for (int i = 0; i < 8; ++i) {
      const int j = j0 + i;
      float s = b1[j];
      const float* wr = w1 + (size_t)j * 256;
      #pragma unroll 4
      for (int k = 0; k < 256; ++k) s += wr[k] * zr[k];
      hid[bb * 128 + j] = fmaxf(s, 0.f);
    }
  }
  __syncthreads();
  {
    const int j0 = (tid & 15) * 8;
    const float* hr = hid + bb * 128;
    float acc[8];
    #pragma unroll
    for (int i = 0; i < 8; ++i) {
      const int j = j0 + i;
      float s = b2[j];
      const float* wr = w2 + (size_t)j * 128;
      #pragma unroll 4
      for (int k = 0; k < 128; ++k) s += wr[k] * hr[k];
      acc[i] = s;
    }
    __syncthreads();                       // z region now dead
    #pragma unroll
    for (int i = 0; i < 8; ++i) mv[bb * 128 + j0 + i] = acc[i];
  }
  __syncthreads();
  {
    const int qt = qt_s[bb];
    float* girow = gi + ((size_t)(bb * NQ_) + qt) * G3_;
    const float* mvr = mv + bb * 128;
    for (int ii = 0; ii < 24; ++ii) {
      const int g = (tid & 15) + 16 * ii;
      float s = 0.f;
      const float* wr = w_ih + (size_t)g * 256;   // first-half columns
      #pragma unroll 4
      for (int k = 0; k < 128; ++k) s += wr[k] * mvr[k];
      girow[g] += s;
    }
  }
  __threadfence();
  __syncthreads();

  // ---- pin w_hh B-fragments in VGPRs for the whole node scan
  const int lane = tid & 31;
  const int wv   = tid >> 5;        // 0..15
  const int mt   = wv >> 3;         // M tile: batches 0-15 / 16-31
  const int ws   = wv & 7;          // 3 N-tiles each
  const int ln15 = lane & 15;
  const int e0   = (lane >> 4) * 8;
  const int g0   = 48 * ws;

  FragBF Bf[12];
  #pragma unroll
  for (int j = 0; j < 3; ++j) {
    const bf16* wp = whh_bf + (size_t)(g0 + 16 * j + ln15) * H_;
    #pragma unroll
    for (int ks = 0; ks < 4; ++ks) {
      Bf[j * 4 + ks].q[0] = *(const u32x4*)(wp + ks * 32 + e0);
      Bf[j * 4 + ks].q[1] = *(const u32x4*)(wp + ks * 32 + e0 + 16);
    }
  }

  const int jb   = (tid & 15) * 8;        // elementwise: 8 hidden units of batch bb
  const int arow = mt * 16 + ln15;        // A-fragment row in hbf
  const int mb   = mt * 16 + (lane >> 4) * 8;
  float hreg[8] = {0.f, 0.f, 0.f, 0.f, 0.f, 0.f, 0.f, 0.f};   // GRU carry (h0=0)

  for (int n = 0; n < NQ_; ++n) {
    // prefetch gi for this node (independent of the WMMAs below)
    const float* gptr = gi + ((size_t)(bb * NQ_) + n) * G3_ + jb;
    float gr[8], gz[8], gn[8];
    #pragma unroll
    for (int i = 0; i < 8; ++i) {
      gr[i] = gptr[i]; gz[i] = gptr[128 + i]; gn[i] = gptr[256 + i];
    }

    // gh = carry @ w_hh.T  (2 M-tiles x 24 N-tiles over 16 waves)
    v8f c0 = {}, c1 = {}, c2 = {};
    #pragma unroll
    for (int ks = 0; ks < 4; ++ks) {
      FragBF a;
      a.q[0] = *(const u32x4*)(hbf + arow * 136 + ks * 32 + e0);
      a.q[1] = *(const u32x4*)(hbf + arow * 136 + ks * 32 + e0 + 16);
      c0 = WMMA_BF16(a.v, Bf[0 * 4 + ks].v, c0);
      c1 = WMMA_BF16(a.v, Bf[1 * 4 + ks].v, c1);
      c2 = WMMA_BF16(a.v, Bf[2 * 4 + ks].v, c2);
    }
    #pragma unroll
    for (int v = 0; v < 8; ++v) {
      float* grow = gh_ls + (mb + v) * 388;
      grow[g0      + ln15] = c0[v] + bhh_s[g0      + ln15];
      grow[g0 + 16 + ln15] = c1[v] + bhh_s[g0 + 16 + ln15];
      grow[g0 + 32 + ln15] = c2[v] + bhh_s[g0 + 32 + ln15];
    }
    __syncthreads();

    // GRU elementwise + outputs
    float yp = 0.f;
    float* hdst = hout + ((size_t)(bb * (T_ + 1) + t + 1) * NQ_ + n) * H_ + jb;
    const float* ghr = gh_ls + bb * 388;
    #pragma unroll
    for (int i = 0; i < 8; ++i) {
      const int j = jb + i;
      const float rg = sigmoidf_(gr[i] + ghr[j]);
      const float zg = sigmoidf_(gz[i] + ghr[128 + j]);
      const float ng = tanhf(gn[i] + rg * ghr[256 + j]);
      const float hv = (1.f - zg) * ng + zg * hreg[i];
      hreg[i] = hv;
      hbf[bb * 136 + j] = (bf16)hv;
      hdst[i] = hv;
      yp += hv * outw_s[j];
    }
    atomicAdd(&y_acc[bb], yp);
    __syncthreads();

    if (tid < 32) {
      yout[((size_t)tid * T_ + t) * NQ_ + n] = sigmoidf_(y_acc[tid] + bias_s[n]);
      y_acc[tid] = 0.f;
    }
  }
}

// ---------------------------------------------------------------------------
extern "C" void kernel_launch(void* const* d_in, const int* in_sizes, int n_in,
                              void* d_out, int out_size, void* d_ws, size_t ws_size,
                              hipStream_t stream) {
  const int*   q    = (const int*)d_in[0];
  const int*   r    = (const int*)d_in[1];
  const float* xe   = (const float*)d_in[2];
  /* d_in[3] = q_emb: algebraically unused (masked out by onehot) */
  const float* ih0  = (const float*)d_in[4];
  const float* w1   = (const float*)d_in[5];
  const float* b1   = (const float*)d_in[6];
  const float* w2   = (const float*)d_in[7];
  const float* b2   = (const float*)d_in[8];
  const float* wih  = (const float*)d_in[9];
  const float* whh  = (const float*)d_in[10];
  const float* bih  = (const float*)d_in[11];
  const float* bhh  = (const float*)d_in[12];
  const float* bias = (const float*)d_in[13];
  const float* ow   = (const float*)d_in[14];

  float* yout = (float*)d_out;                          // [B, T, NQ]
  float* hout = yout + (size_t)B_ * T_ * NQ_;           // [B, T+1, NQ, H]

  // workspace: gi (6 MB f32) | whh_bf (96 KB) | wh_bf (96 KB)
  float* gi   = (float*)d_ws;
  bf16*  whhb = (bf16*)((char*)d_ws + 6291456);
  bf16*  whb  = (bf16*)((char*)d_ws + 6389760);

  gkt_init<<<2048, 256, 0, stream>>>(ih0, wih, whh, hout, whhb, whb);
  for (int t = 0; t < T_; ++t) {
    gkt_gemm_gi<<<256, 256, 0, stream>>>(hout, whb, bih, gi, t);
    gkt_scan<<<1, 512, 0, stream>>>(q, r, xe, w1, b1, w2, b2, wih, bhh,
                                    bias, ow, whhb, gi, hout, yout, t);
  }
}
